// PureGNN_57767310131228
// MI455X (gfx1250) — compile-verified, matrix operand order checked by reading, not run on hardware
//
#include <hip/hip_runtime.h>
#include <hip/hip_bf16.h>

// ---------------------------------------------------------------------------
// GCN forward for MI455X (gfx1250, wave32, WMMA).
//   - GEMMs: v_wmma_f32_16x16x32_f16, one 16x16 output tile per wave.
//   - Message passing: float4 gather + global fp32 atomics (bandwidth bound).
//   - Pooling: atomic sum + bit-pattern atomic max (activations are >= 0).
// ---------------------------------------------------------------------------

#define NN 100000
#define NE 800000
#define NG 64
#define NFEAT 6
#define NCLS 2
#define EPSV 1e-5f

typedef _Float16 h8  __attribute__((ext_vector_type(8)));
typedef _Float16 h16 __attribute__((ext_vector_type(16)));
typedef float    f8  __attribute__((ext_vector_type(8)));

// ---------------------------------------------------------------- utilities
__global__ void zero_f32_k(float* __restrict__ p, long long n) {
  long long i = (long long)blockIdx.x * blockDim.x + threadIdx.x;
  if (i < n) p[i] = 0.0f;
}

// deg[dst] += 1 over edges
__global__ void deg_k(const long long* __restrict__ ei, float* __restrict__ deg) {
  int e = blockIdx.x * blockDim.x + threadIdx.x;
  if (e < NE) atomicAdd(&deg[(int)ei[NE + e]], 1.0f);
}

// dis = (deg+1)^-0.5 ; selfw = (deg+1)^-1
__global__ void dis_k(const float* __restrict__ deg, float* __restrict__ dis,
                      float* __restrict__ selfw) {
  int n = blockIdx.x * blockDim.x + threadIdx.x;
  if (n >= NN) return;
  float d = deg[n] + 1.0f;
  float r = rsqrtf(d);
  dis[n] = r;
  selfw[n] = r * r;
}

// W [din,dout] fp32 row-major  ->  WT16 [dout, KP] fp16 (B^T, K-contiguous, zero pad)
__global__ void pack_wt_k(const float* __restrict__ W, _Float16* __restrict__ WT,
                          int din, int dout, int KP) {
  int i = blockIdx.x * blockDim.x + threadIdx.x;
  if (i >= dout * KP) return;
  int c = i / KP, k = i % KP;
  WT[i] = (k < din) ? (_Float16)W[(size_t)k * dout + c] : (_Float16)0.0f;
}

// node_features [N,6] -> x16 [N,32] fp16 zero padded
__global__ void pack_x0_k(const float* __restrict__ nf, _Float16* __restrict__ x16) {
  long long i = (long long)blockIdx.x * blockDim.x + threadIdx.x;
  if (i >= (long long)NN * 32) return;
  int n = (int)(i >> 5), k = (int)(i & 31);
  x16[i] = (k < NFEAT) ? (_Float16)nf[(size_t)n * NFEAT + k] : (_Float16)0.0f;
}

// ---------------------------------------------------------------- WMMA GEMM
// C[M,Ncols] = A[M,KP] @ B, with B supplied as BT[Ncols,KP] (both fp16).
// One wave per 16x16 output tile. KP multiple of 32, M and Ncols multiples of 16.
// A-frag per ISA 7.12.2 (16-bit A 16x32): lanes 0-15 hold K{0..7,16..23},
// lanes 16-31 hold K{8..15,24..31}; B mirrors with N in place of M.
__global__ void gemm_wmma_k(const _Float16* __restrict__ A,
                            const _Float16* __restrict__ BT,
                            float* __restrict__ C, _Float16* __restrict__ C16,
                            const float* __restrict__ bias, int relu,
                            int M, int KP, int Ncols) {
  int wave = blockIdx.x * (blockDim.x >> 5) + (threadIdx.x >> 5);
  int lane = threadIdx.x & 31;
  int Ntiles = Ncols >> 4;
  int mt = wave / Ntiles;
  int nt = wave % Ntiles;
  if (mt * 16 >= M) return;                       // wave-uniform: EXEC stays all-1s

  int r    = lane & 15;
  int koff = (lane < 16) ? 0 : 8;
  const _Float16* arow = A  + (size_t)(mt * 16 + r) * KP;
  const _Float16* brow = BT + (size_t)(nt * 16 + r) * KP;

  f8 acc = {};
  for (int k0 = 0; k0 < KP; k0 += 32) {
    union { h16 v; h8 p[2]; } a, b;
    a.p[0] = *(const h8*)(arow + k0 + koff);
    a.p[1] = *(const h8*)(arow + k0 + koff + 16);
    b.p[0] = *(const h8*)(brow + k0 + koff);
    b.p[1] = *(const h8*)(brow + k0 + koff + 16);
    acc = __builtin_amdgcn_wmma_f32_16x16x32_f16(
        /*neg_a=*/false, a.v, /*neg_b=*/false, b.v,
        /*c_mod=*/(short)0, acc, /*reuse_a=*/false, /*reuse_b=*/false);
  }

  // C/D layout: VGPR v -> row = mt*16 + v + (lane>=16 ? 8 : 0), col = nt*16 + lane%16
  int row0 = mt * 16 + ((lane < 16) ? 0 : 8);
  int col  = nt * 16 + r;
  float bv = bias ? bias[col] : 0.0f;
#pragma unroll
  for (int v = 0; v < 8; ++v) {
    float x = acc[v] + bv;
    if (relu) x = fmaxf(x, 0.0f);
    size_t o = (size_t)(row0 + v) * Ncols + col;
    C[o] = x;
    if (C16) C16[o] = (_Float16)x;
  }
}

// ------------------------------------------------------- edge scatter (atomics)
// agg[dst] += h[src] * dis[src]*dis[dst]; 4 features per thread.
__global__ void edge_msg_k(const long long* __restrict__ ei,
                           const float* __restrict__ h,
                           const float* __restrict__ dis,
                           float* __restrict__ agg, int dout) {
  long long idx = (long long)blockIdx.x * blockDim.x + threadIdx.x;
  int fv = dout >> 2;
  long long total = (long long)NE * fv;
  if (idx >= total) return;
  int e  = (int)(idx / fv);
  int f4 = (int)(idx % fv) << 2;
  int src = (int)ei[e];
  int dst = (int)ei[NE + e];
  float nrm = dis[src] * dis[dst];
  const float4 hv = *(const float4*)(h + (size_t)src * dout + f4);
  float* ap = agg + (size_t)dst * dout + f4;
  atomicAdd(ap + 0, hv.x * nrm);
  atomicAdd(ap + 1, hv.y * nrm);
  atomicAdd(ap + 2, hv.z * nrm);
  atomicAdd(ap + 3, hv.w * nrm);
}

// ---------------------------------------------- fused self-loop + bias + BN + ReLU
// agg <- relu(bn(agg + h*selfw + b)) in place; also emit fp16 copy for next GEMM.
__global__ void gcn_epi_k(float* __restrict__ agg, const float* __restrict__ h,
                          const float* __restrict__ selfw,
                          const float* __restrict__ b, const float* __restrict__ g,
                          const float* __restrict__ be, const float* __restrict__ rm,
                          const float* __restrict__ rv,
                          _Float16* __restrict__ x16, int dout) {
  long long idx = (long long)blockIdx.x * blockDim.x + threadIdx.x;
  long long total = (long long)NN * dout;
  if (idx >= total) return;
  int n = (int)(idx / dout);
  int f = (int)(idx % dout);
  float v = agg[idx] + h[idx] * selfw[n] + b[f];
  v = (v - rm[f]) * rsqrtf(rv[f] + EPSV) * g[f] + be[f];
  v = fmaxf(v, 0.0f);
  agg[idx] = v;
  x16[idx] = (_Float16)v;
}

// ---------------------------------------------------------------- pooling
__global__ void pool_cnt_k(const long long* __restrict__ batch, float* __restrict__ cnt) {
  int n = blockIdx.x * blockDim.x + threadIdx.x;
  if (n < NN) atomicAdd(&cnt[(int)batch[n]], 1.0f);
}

__global__ void pool_sm_k(const long long* __restrict__ batch,
                          const float* __restrict__ x,
                          float* __restrict__ sum, unsigned* __restrict__ mx) {
  long long idx = (long long)blockIdx.x * blockDim.x + threadIdx.x;
  long long total = (long long)NN * 512;
  if (idx >= total) return;
  int n = (int)(idx >> 9);
  int f = (int)(idx & 511);
  int gi = (int)batch[n] * 512 + f;
  float v = x[idx];
  atomicAdd(&sum[gi], v);
  atomicMax(&mx[gi], __float_as_uint(v));   // v >= 0 post-ReLU: bit order == value order
}

__global__ void hcat_k(const float* __restrict__ sum, const unsigned* __restrict__ mx,
                       const float* __restrict__ cnt,
                       float* __restrict__ hc, _Float16* __restrict__ hc16) {
  int idx = blockIdx.x * blockDim.x + threadIdx.x;
  if (idx >= NG * 1024) return;
  int g = idx >> 10, f = idx & 1023;
  float v = (f < 512) ? sum[g * 512 + f] / fmaxf(cnt[g], 1.0f)
                      : __uint_as_float(mx[g * 512 + (f - 512)]);
  hc[idx] = v;
  hc16[idx] = (_Float16)v;
}

// ------------------------------------------------------------ final 128 -> 2 head
__global__ void head_k(const float* __restrict__ m2, const float* __restrict__ W,
                       const float* __restrict__ b, float* __restrict__ out) {
  int idx = blockIdx.x * blockDim.x + threadIdx.x;
  if (idx >= NG * NCLS) return;
  int g = idx / NCLS, c = idx % NCLS;
  float s = b[c];
  for (int k = 0; k < 128; ++k) s += m2[g * 128 + k] * W[k * NCLS + c];
  out[idx] = s;
}

// ---------------------------------------------------------------------------
extern "C" void kernel_launch(void* const* d_in, const int* in_sizes, int n_in,
                              void* d_out, int out_size, void* d_ws, size_t ws_size,
                              hipStream_t stream) {
  (void)in_sizes; (void)n_in; (void)out_size; (void)ws_size;

  const float*     nf    = (const float*)d_in[0];
  const long long* ei    = (const long long*)d_in[1];
  const long long* batch = (const long long*)d_in[2];
  // layer params: W,b,g,be,rm,rv starting at index 3, 6 per layer
  const float* Wl[4];  const float* bl[4]; const float* gl[4];
  const float* bel[4]; const float* rml[4]; const float* rvl[4];
  for (int l = 0; l < 4; ++l) {
    int base = 3 + 6 * l;
    Wl[l]  = (const float*)d_in[base + 0];
    bl[l]  = (const float*)d_in[base + 1];
    gl[l]  = (const float*)d_in[base + 2];
    bel[l] = (const float*)d_in[base + 3];
    rml[l] = (const float*)d_in[base + 4];
    rvl[l] = (const float*)d_in[base + 5];
  }
  const float* Wf1 = (const float*)d_in[27]; const float* bf1 = (const float*)d_in[28];
  const float* Wf2 = (const float*)d_in[29]; const float* bf2 = (const float*)d_in[30];
  const float* Wf3 = (const float*)d_in[31]; const float* bf3 = (const float*)d_in[32];
  float* out = (float*)d_out;

  // ---- workspace carve-up (256B aligned) ----
  char* base = (char*)d_ws;
  size_t off = 0;
  auto carve = [&](size_t bytes) -> char* {
    char* p = base + off;
    off += (bytes + 255) & ~(size_t)255;
    return p;
  };
  float*     deg   = (float*)carve((size_t)NN * 4);
  float*     dis   = (float*)carve((size_t)NN * 4);
  float*     selfw = (float*)carve((size_t)NN * 4);
  float*     h     = (float*)carve((size_t)NN * 512 * 4);
  float*     agg   = (float*)carve((size_t)NN * 512 * 4);
  _Float16*  x16   = (_Float16*)carve((size_t)NN * 512 * 2);
  _Float16*  wt16  = (_Float16*)carve((size_t)1024 * 512 * 2);
  float*     cnt   = (float*)carve((size_t)NG * 4);
  float*     psum  = (float*)carve((size_t)NG * 512 * 4);
  unsigned*  pmax  = (unsigned*)carve((size_t)NG * 512 * 4);
  float*     hcat  = (float*)carve((size_t)NG * 1024 * 4);
  _Float16*  hc16  = (_Float16*)carve((size_t)NG * 1024 * 2);
  float*     m1    = (float*)carve((size_t)NG * 256 * 4);
  _Float16*  m1h   = (_Float16*)carve((size_t)NG * 256 * 2);
  float*     m2    = (float*)carve((size_t)NG * 128 * 4);
  _Float16*  m2h   = (_Float16*)carve((size_t)NG * 128 * 2);

  const int T = 256;
  auto blocks = [&](long long n) { return (unsigned)((n + T - 1) / T); };

  // ---- degree / normalization ----
  zero_f32_k<<<blocks(NN), T, 0, stream>>>(deg, NN);
  deg_k<<<blocks(NE), T, 0, stream>>>(ei, deg);
  dis_k<<<blocks(NN), T, 0, stream>>>(deg, dis, selfw);

  // ---- initial fp16 features, padded 6 -> 32 ----
  pack_x0_k<<<blocks((long long)NN * 32), T, 0, stream>>>(nf, x16);

  // ---- GCN layers ----
  const int dins[4]  = {NFEAT, 64, 128, 256};
  const int KPs[4]   = {32,    64, 128, 256};
  const int douts[4] = {64,   128, 256, 512};
  for (int l = 0; l < 4; ++l) {
    int din = dins[l], KP = KPs[l], dout = douts[l];
    pack_wt_k<<<blocks((long long)dout * KP), T, 0, stream>>>(Wl[l], wt16, din, dout, KP);
    long long tiles = (long long)(NN / 16) * (dout / 16);
    gemm_wmma_k<<<blocks(tiles * 32), 128, 0, stream>>>(
        x16, wt16, h, (_Float16*)nullptr, (const float*)nullptr, 0, NN, KP, dout);
    zero_f32_k<<<blocks((long long)NN * dout), T, 0, stream>>>(agg, (long long)NN * dout);
    edge_msg_k<<<blocks((long long)NE * (dout >> 2)), T, 0, stream>>>(ei, h, dis, agg, dout);
    gcn_epi_k<<<blocks((long long)NN * dout), T, 0, stream>>>(
        agg, h, selfw, bl[l], gl[l], bel[l], rml[l], rvl[l], x16, dout);
  }

  // ---- pooling (agg holds layer-4 activations, fp32) ----
  zero_f32_k<<<blocks(NG), T, 0, stream>>>(cnt, NG);
  zero_f32_k<<<blocks((long long)NG * 512), T, 0, stream>>>(psum, (long long)NG * 512);
  zero_f32_k<<<blocks((long long)NG * 512), T, 0, stream>>>((float*)pmax, (long long)NG * 512);
  pool_cnt_k<<<blocks(NN), T, 0, stream>>>(batch, cnt);
  pool_sm_k<<<blocks((long long)NN * 512), T, 0, stream>>>(batch, agg, psum, pmax);
  hcat_k<<<blocks((long long)NG * 1024), T, 0, stream>>>(psum, pmax, cnt, hcat, hc16);

  // ---- MLP head (WMMA again; M = 64 rows) ----
  pack_wt_k<<<blocks((long long)256 * 1024), T, 0, stream>>>(Wf1, wt16, 1024, 256, 1024);
  gemm_wmma_k<<<blocks((long long)(NG / 16) * (256 / 16) * 32), 128, 0, stream>>>(
      hc16, wt16, m1, m1h, bf1, 1, NG, 1024, 256);
  pack_wt_k<<<blocks((long long)128 * 256), T, 0, stream>>>(Wf2, wt16, 256, 128, 256);
  gemm_wmma_k<<<blocks((long long)(NG / 16) * (128 / 16) * 32), 128, 0, stream>>>(
      m1h, wt16, m2, m2h, bf2, 1, NG, 256, 128);
  head_k<<<blocks(NG * NCLS), T, 0, stream>>>(m2, Wf3, bf3, out);
}
